// UtteranceDecoder_67877663146254
// MI455X (gfx1250) — matrix operand order repaired and checked by reading, not compiled, not run on hardware
//
#include <hip/hip_runtime.h>
#include <hip/hip_bf16.h>

typedef __attribute__((ext_vector_type(16))) _Float16 v16h;
typedef __attribute__((ext_vector_type(8)))  _Float16 v8h;
typedef __attribute__((ext_vector_type(8)))  float    v8f;

#define Dm   1536
#define DFFm 2048
#define Hm   8
#define HDm  192      // D / H
#define Bm   16
#define Lm   64
#define Um   64
#define ROWS 1024     // B * L
#define NLAY 2

__device__ __forceinline__ v8f wmma_f16(v16h a, v16h b, v8f c) {
    // D = A(16x32 f16) * B(32x16 f16) + C(16x16 f32)
    return __builtin_amdgcn_wmma_f32_16x16x32_f16(
        /*neg_a=*/false, a, /*neg_b=*/false, b,
        /*c_mod=*/(short)0, c, /*reuse_a=*/false, /*reuse_b=*/false);
}

// ---------------------------------------------------------------------------
// fp32 -> f16 conversion, 8 elements/thread, b128 in / b128 out
// ---------------------------------------------------------------------------
__global__ void cvt_f32_f16_kernel(const float* __restrict__ src,
                                   _Float16* __restrict__ dst, int n) {
    const int i = (blockIdx.x * 256 + threadIdx.x) * 8;
    if (i + 8 <= n) {
        const float4 f0 = *(const float4*)(src + i);
        const float4 f1 = *(const float4*)(src + i + 4);
        v8h o;
        o[0] = (_Float16)f0.x; o[1] = (_Float16)f0.y;
        o[2] = (_Float16)f0.z; o[3] = (_Float16)f0.w;
        o[4] = (_Float16)f1.x; o[5] = (_Float16)f1.y;
        o[6] = (_Float16)f1.z; o[7] = (_Float16)f1.w;
        *(v8h*)(dst + i) = o;
    } else {
        for (int j = i; j < n; ++j) dst[j] = (_Float16)src[j];
    }
}

// ---------------------------------------------------------------------------
// GEMM: Y[m, n] = relu?( sum_k A[m,k] * W[n,k] + bias[n] )
// A: M x K f16 row-major. W: N x K f16 row-major. Y fp32, row stride ldo.
// One wave -> 16(M) x 64(N) tile; 8 waves / 256-thread block.
// M % 16 == 0, N % 64 == 0, K % 32 == 0 (true for this model).
// ---------------------------------------------------------------------------
__global__ void gemm_wmma_f16_kernel(const _Float16* __restrict__ A,
                                     const _Float16* __restrict__ W,
                                     const float* __restrict__ bias,
                                     float* __restrict__ Y,
                                     int M, int N, int K, int ldo, int relu) {
    const int lane = threadIdx.x & 31;
    const int wave = threadIdx.x >> 5;
    const int tile = blockIdx.x * (blockDim.x >> 5) + wave;
    const int tilesN = N >> 6;
    const int tm = tile / tilesN;
    const int tn = tile % tilesN;
    if (tm * 16 >= M) return;              // wave-uniform

    const int nlo = lane & 15;
    const int koA = (lane >= 16) ? 16 : 0; // A frag: 16 contiguous K starting at koA
    const int koB = (lane >= 16) ? 16 : 0; // B frag: 16 contiguous K starting at koB

    const _Float16* __restrict__ arow = A + (size_t)(tm * 16 + nlo) * K + koA;
    const _Float16* __restrict__ w0 = W + (size_t)(tn * 64 + nlo +  0) * K + koB;
    const _Float16* __restrict__ w1 = W + (size_t)(tn * 64 + nlo + 16) * K + koB;
    const _Float16* __restrict__ w2 = W + (size_t)(tn * 64 + nlo + 32) * K + koB;
    const _Float16* __restrict__ w3 = W + (size_t)(tn * 64 + nlo + 48) * K + koB;

    v8f acc0 = {}, acc1 = {}, acc2 = {}, acc3 = {};

    for (int k0 = 0; k0 < K; k0 += 32) {
        // issue all fragment loads up front (pure f16, b128-friendly)
        const v8h alo = *(const v8h*)(arow + k0);
        const v8h ahi = *(const v8h*)(arow + k0 + 8);
        const v16h b0 = *(const v16h*)(w0 + k0);
        const v16h b1 = *(const v16h*)(w1 + k0);
        const v16h b2 = *(const v16h*)(w2 + k0);
        const v16h b3 = *(const v16h*)(w3 + k0);

        if (k0 + 256 < K) {                 // ~8 K-steps ahead
            __builtin_prefetch(arow + k0 + 256, 0, 3);
            __builtin_prefetch(w0 + k0 + 256, 0, 3);
            __builtin_prefetch(w1 + k0 + 256, 0, 3);
            __builtin_prefetch(w2 + k0 + 256, 0, 3);
            __builtin_prefetch(w3 + k0 + 256, 0, 3);
        }

        v16h a;
#pragma unroll
        for (int i = 0; i < 8; ++i) { a[i] = alo[i]; a[8 + i] = ahi[i]; }

        acc0 = wmma_f16(a, b0, acc0);
        acc1 = wmma_f16(a, b1, acc1);
        acc2 = wmma_f16(a, b2, acc2);
        acc3 = wmma_f16(a, b3, acc3);
    }

    // C/D layout: VGPR r, lanes 0-15 -> (M=r, N=lane); lanes 16-31 -> (M=r+8, N=lane-16)
    const int mhi = (lane >> 4) << 3;
    const int cb  = tn * 64;
#pragma unroll
    for (int r = 0; r < 8; ++r) {
        const int m = tm * 16 + r + mhi;
        float* __restrict__ yrow = Y + (size_t)m * ldo;
        float v0 = acc0[r] + bias[cb + nlo +  0];
        float v1 = acc1[r] + bias[cb + nlo + 16];
        float v2 = acc2[r] + bias[cb + nlo + 32];
        float v3 = acc3[r] + bias[cb + nlo + 48];
        if (relu) {
            v0 = fmaxf(v0, 0.f); v1 = fmaxf(v1, 0.f);
            v2 = fmaxf(v2, 0.f); v3 = fmaxf(v3, 0.f);
        }
        yrow[cb + nlo +  0] = v0;
        yrow[cb + nlo + 16] = v1;
        yrow[cb + nlo + 32] = v2;
        yrow[cb + nlo + 48] = v3;
    }
}

// ---------------------------------------------------------------------------
// Pooling: logits[u] = x[b,l,u,:] . pw + pb ; att = softmax over U;
// pooled[d] = sum_u att[u] x[b,l,u,d]; h = mem = pooled + pos[l].
// One 256-thread block per (b,l).
// ---------------------------------------------------------------------------
__global__ void pooler_kernel(const float* __restrict__ x,
                              const float* __restrict__ pw,
                              const float* __restrict__ pb,
                              const float* __restrict__ pos,
                              float* __restrict__ hout,
                              float* __restrict__ mout) {
    const int bl = blockIdx.x;            // b*L + l
    const int l  = bl & (Lm - 1);
    const int t  = threadIdx.x;           // 256
    __shared__ float lg[256];
    __shared__ float att[Um];

    const float* __restrict__ xb = x + (size_t)bl * Um * Dm;
    const int u = t & 63, part = t >> 6;  // 4 partial sums per u
    {
        const float4* __restrict__ xu  =
            (const float4*)(xb + (size_t)u * Dm) + part * (Dm / 16);
        const float4* __restrict__ pwp = (const float4*)pw + part * (Dm / 16);
        float s = 0.f;
        for (int i = 0; i < Dm / 16; ++i) {
            const float4 xv = xu[i], pv = pwp[i];
            s += xv.x * pv.x + xv.y * pv.y + xv.z * pv.z + xv.w * pv.w;
        }
        lg[t] = s;
    }
    __syncthreads();
    if (t < Um) lg[t] = lg[t] + lg[t + 64] + lg[t + 128] + lg[t + 192] + pb[0];
    __syncthreads();
    if (t < Um) {
        float mx = -1e30f;
        for (int j = 0; j < Um; ++j) mx = fmaxf(mx, lg[j]);
        float sum = 0.f;
        for (int j = 0; j < Um; ++j) sum += __expf(lg[j] - mx);
        att[t] = __expf(lg[t] - mx) / sum;
    }
    __syncthreads();
#pragma unroll
    for (int c = 0; c < Dm / 256; ++c) {
        const int d = t + 256 * c;
        float acc = 0.f;
        for (int uu = 0; uu < Um; ++uu) acc += att[uu] * xb[(size_t)uu * Dm + d];
        const float v = acc + pos[l * Dm + d];
        hout[(size_t)bl * Dm + d] = v;
        mout[(size_t)bl * Dm + d] = v;
    }
}

// ---------------------------------------------------------------------------
// Attention for one (b, head, qpos) per 64-thread block.
// ---------------------------------------------------------------------------
__global__ void attention_kernel(const float* __restrict__ Q, int ldq,
                                 const float* __restrict__ Kx, int ldk,
                                 const float* __restrict__ V, int ldv,
                                 float* __restrict__ O, int ldo,
                                 int causal) {
    const int idx  = blockIdx.x;
    const int qpos = idx & (Lm - 1);
    const int head = (idx >> 6) & (Hm - 1);
    const int b    = idx >> 9;            // / (L*H)
    const int t    = threadIdx.x;         // 64 = one key per thread
    const int rq   = b * Lm + qpos;
    const int rk   = b * Lm + t;

    __shared__ float sc[Lm];

    const float4* __restrict__ q4 = (const float4*)(Q + (size_t)rq * ldq + head * HDm);
    const float4* __restrict__ k4 = (const float4*)(Kx + (size_t)rk * ldk + head * HDm);

    float dot = 0.f;
    for (int d = 0; d < HDm / 4; ++d) {
        const float4 qv = q4[d], kv = k4[d];
        dot += qv.x * kv.x + qv.y * kv.y + qv.z * kv.z + qv.w * kv.w;
    }
    dot *= 0.07216878364870323f;          // 1/sqrt(192)
    if (causal && t > qpos) dot = -1e30f;
    sc[t] = dot;
    __syncthreads();

    float mx = -1e30f;
    for (int j = 0; j < Lm; ++j) mx = fmaxf(mx, sc[j]);
    float sum = 0.f;
    for (int j = 0; j < Lm; ++j) sum += __expf(sc[j] - mx);
    const float p = __expf(sc[t] - mx) / sum;
    __syncthreads();
    sc[t] = p;
    __syncthreads();

#pragma unroll
    for (int c = 0; c < HDm / Lm; ++c) {  // 3 dims per thread, coalesced
        const int d = t + Lm * c;
        float acc = 0.f;
        for (int j = 0; j < Lm; ++j)
            acc += sc[j] * V[(size_t)(b * Lm + j) * ldv + head * HDm + d];
        O[(size_t)rq * ldo + head * HDm + d] = acc;
    }
}

// ---------------------------------------------------------------------------
// Y = LayerNorm(X + R) * w + b   (one 256-thread block per row; Y may alias R)
// ---------------------------------------------------------------------------
__global__ void add_ln_kernel(const float* __restrict__ X,
                              const float* __restrict__ R,
                              const float* __restrict__ w,
                              const float* __restrict__ bvec,
                              float* __restrict__ Y) {
    const int row = blockIdx.x;
    const int t   = threadIdx.x;          // 256
    __shared__ float red[256];
    float vals[Dm / 256];

    float s = 0.f;
#pragma unroll
    for (int c = 0; c < Dm / 256; ++c) {
        const int d = t + 256 * c;
        const float v = X[(size_t)row * Dm + d] + R[(size_t)row * Dm + d];
        vals[c] = v;
        s += v;
    }
    red[t] = s; __syncthreads();
    for (int off = 128; off > 0; off >>= 1) {
        if (t < off) red[t] += red[t + off];
        __syncthreads();
    }
    const float mean = red[0] * (1.f / Dm);
    __syncthreads();

    float sq = 0.f;
#pragma unroll
    for (int c = 0; c < Dm / 256; ++c) {
        const float dv = vals[c] - mean;
        sq += dv * dv;
    }
    red[t] = sq; __syncthreads();
    for (int off = 128; off > 0; off >>= 1) {
        if (t < off) red[t] += red[t + off];
        __syncthreads();
    }
    const float inv = rsqrtf(red[0] * (1.f / Dm) + 1e-5f);

#pragma unroll
    for (int c = 0; c < Dm / 256; ++c) {
        const int d = t + 256 * c;
        Y[(size_t)row * Dm + d] = (vals[c] - mean) * inv * w[d] + bvec[d];
    }
}

// out (B, L-1, D) <- h rows with l < L-1
__global__ void extract_kernel(const float* __restrict__ h, float* __restrict__ out) {
    const int idx = blockIdx.x * 256 + threadIdx.x;
    const int total = Bm * (Lm - 1) * Dm;
    if (idx >= total) return;
    const int d = idx % Dm;
    const int l = (idx / Dm) % (Lm - 1);
    const int b = idx / (Dm * (Lm - 1));
    out[idx] = h[((size_t)(b * Lm + l)) * Dm + d];
}

// ---------------------------------------------------------------------------
static inline void launch_gemm(const float* A, const float* W, const float* bias,
                               float* Y, int M, int N, int K, int ldo, int relu,
                               _Float16* af16, _Float16* wf16, hipStream_t stream) {
    const int na = M * K, nw = N * K;
    cvt_f32_f16_kernel<<<(na + 2047) / 2048, 256, 0, stream>>>(A, af16, na);
    cvt_f32_f16_kernel<<<(nw + 2047) / 2048, 256, 0, stream>>>(W, wf16, nw);
    const int tiles  = (M / 16) * (N / 64);
    const int blocks = tiles / 8;   // 8 waves (256 threads) per block; exact here
    gemm_wmma_f16_kernel<<<blocks, 256, 0, stream>>>(af16, wf16, bias, Y, M, N, K, ldo, relu);
}

extern "C" void kernel_launch(void* const* d_in, const int* in_sizes, int n_in,
                              void* d_out, int out_size, void* d_ws, size_t ws_size,
                              hipStream_t stream) {
    const float* x        = (const float*)d_in[0];
    const float* pooler_w = (const float*)d_in[1];
    const float* pooler_b = (const float*)d_in[2];
    const float* pos      = (const float*)d_in[3];
    const float* sa_qkv_w = (const float*)d_in[4];
    const float* sa_qkv_b = (const float*)d_in[5];
    const float* sa_out_w = (const float*)d_in[6];
    const float* sa_out_b = (const float*)d_in[7];
    const float* ca_qkv_w = (const float*)d_in[8];
    const float* ca_qkv_b = (const float*)d_in[9];
    const float* ca_out_w = (const float*)d_in[10];
    const float* ca_out_b = (const float*)d_in[11];
    const float* ff1_w    = (const float*)d_in[12];
    const float* ff1_b    = (const float*)d_in[13];
    const float* ff2_w    = (const float*)d_in[14];
    const float* ff2_b    = (const float*)d_in[15];
    const float* ln1_w    = (const float*)d_in[16];
    const float* ln1_b    = (const float*)d_in[17];
    const float* ln2_w    = (const float*)d_in[18];
    const float* ln2_b    = (const float*)d_in[19];
    const float* ln3_w    = (const float*)d_in[20];
    const float* ln3_b    = (const float*)d_in[21];

    float* ws   = (float*)d_ws;
    float* hbuf = ws;                                  // ROWS * D
    float* mem  = hbuf + (size_t)ROWS * Dm;            // ROWS * D
    float* qkv  = mem  + (size_t)ROWS * Dm;            // ROWS * 3D
    float* abuf = qkv  + (size_t)ROWS * 3 * Dm;        // ROWS * D
    float* ffb  = abuf + (size_t)ROWS * Dm;            // ROWS * DFF
    float* tbuf = ffb  + (size_t)ROWS * DFFm;          // ROWS * D
    _Float16* af16 = (_Float16*)(tbuf + (size_t)ROWS * Dm);   // ROWS * DFF halves (max A)
    _Float16* wf16 = af16 + (size_t)ROWS * DFFm;              // 3D * D halves (max W)
    (void)ws_size; (void)in_sizes; (void)n_in; (void)out_size;

    // 1) pooling + positional embed -> h, memory
    pooler_kernel<<<ROWS, 256, 0, stream>>>(x, pooler_w, pooler_b, pos, hbuf, mem);

    const int attn_blocks = Bm * Hm * Lm;   // 8192

    for (int i = 0; i < NLAY; ++i) {
        const float* sqw = sa_qkv_w + (size_t)i * 3 * Dm * Dm;
        const float* sqb = sa_qkv_b + (size_t)i * 3 * Dm;
        const float* sow = sa_out_w + (size_t)i * Dm * Dm;
        const float* sob = sa_out_b + (size_t)i * Dm;
        const float* cqw = ca_qkv_w + (size_t)i * 3 * Dm * Dm;
        const float* cqb = ca_qkv_b + (size_t)i * 3 * Dm;
        const float* cow = ca_out_w + (size_t)i * Dm * Dm;
        const float* cob = ca_out_b + (size_t)i * Dm;
        const float* f1w = ff1_w + (size_t)i * DFFm * Dm;
        const float* f1b = ff1_b + (size_t)i * DFFm;
        const float* f2w = ff2_w + (size_t)i * Dm * DFFm;
        const float* f2b = ff2_b + (size_t)i * Dm;

        // ---- self attention ----
        launch_gemm(hbuf, sqw, sqb, qkv, ROWS, 3 * Dm, Dm, 3 * Dm, 0, af16, wf16, stream);
        attention_kernel<<<attn_blocks, Lm, 0, stream>>>(
            qkv, 3 * Dm, qkv + Dm, 3 * Dm, qkv + 2 * Dm, 3 * Dm,
            abuf, Dm, /*causal=*/1);
        launch_gemm(abuf, sow, sob, tbuf, ROWS, Dm, Dm, Dm, 0, af16, wf16, stream);
        add_ln_kernel<<<ROWS, 256, 0, stream>>>(
            tbuf, hbuf, ln1_w + (size_t)i * Dm, ln1_b + (size_t)i * Dm, hbuf);

        // ---- cross attention: q from h, k/v from memory ----
        launch_gemm(hbuf, cqw, cqb, qkv, ROWS, Dm, Dm, 3 * Dm, 0, af16, wf16, stream);
        launch_gemm(mem, cqw + (size_t)Dm * Dm, cqb + Dm, qkv + Dm,
                    ROWS, 2 * Dm, Dm, 3 * Dm, 0, af16, wf16, stream);
        attention_kernel<<<attn_blocks, Lm, 0, stream>>>(
            qkv, 3 * Dm, qkv + Dm, 3 * Dm, qkv + 2 * Dm, 3 * Dm,
            abuf, Dm, /*causal=*/0);
        launch_gemm(abuf, cow, cob, tbuf, ROWS, Dm, Dm, Dm, 0, af16, wf16, stream);
        add_ln_kernel<<<ROWS, 256, 0, stream>>>(
            tbuf, hbuf, ln2_w + (size_t)i * Dm, ln2_b + (size_t)i * Dm, hbuf);

        // ---- feed forward ----
        launch_gemm(hbuf, f1w, f1b, ffb, ROWS, DFFm, Dm, DFFm, /*relu=*/1, af16, wf16, stream);
        launch_gemm(ffb, f2w, f2b, tbuf, ROWS, Dm, DFFm, Dm, 0, af16, wf16, stream);
        add_ln_kernel<<<ROWS, 256, 0, stream>>>(
            tbuf, hbuf, ln3_w + (size_t)i * Dm, ln3_b + (size_t)i * Dm, hbuf);
    }

    // 2) output: h[:, :L-1, :]
    const int total = Bm * (Lm - 1) * Dm;
    extract_kernel<<<(total + 255) / 256, 256, 0, stream>>>(hbuf, (float*)d_out);
}
